// MinGRU_84902913507469
// MI455X (gfx1250) — compile-verified, matrix operand order checked by reading, not run on hardware
//
#include <hip/hip_runtime.h>
#include <math.h>

// Problem constants (B,S,T,D) = (16, 512, 16, 128)
#define BB 16
#define SS 512
#define TT 16
#define DD 128
#define MM (BB * SS * TT) /* 131072 GEMM rows */
#define LN_EPS 1e-5f

typedef __attribute__((ext_vector_type(16))) _Float16 v16h;
typedef __attribute__((ext_vector_type(8)))  _Float16 v8h;
typedef __attribute__((ext_vector_type(4)))  _Float16 v4h;
typedef __attribute__((ext_vector_type(8)))  float    v8f;

union V16H { v16h v; v8h h[2]; };

__device__ __forceinline__ float softplusf(float x) {
  // log(1+exp(x)), numerically stable
  return fmaxf(x, 0.0f) + log1pf(expf(-fabsf(x)));
}

// ---------------------------------------------------------------------------
// Kernel 1: pack W (DD x 2DD, f32 row-major) into f16 in the wave32 16-bit
// B-matrix WMMA layout. Wt element for (ntile j, kstep kk, lane, half h):
//   N = j*16 + (lane&15)
//   K = kk*32 + h + (h>=8 ? 8 : 0) + (lane>=16 ? 8 : 0)
// stored at Wt[((j*4+kk)*32 + lane)*16 + h]  (== flat tid).
// ---------------------------------------------------------------------------
__global__ void wpack_kernel(const float* __restrict__ W, _Float16* __restrict__ Wt) {
  int tid  = blockIdx.x * blockDim.x + threadIdx.x;   // 0..32767
  int h    = tid & 15;
  int lane = (tid >> 4) & 31;
  int kk   = (tid >> 9) & 3;
  int j    = tid >> 11;                               // 0..15
  int K = kk * 32 + h + ((h >= 8) ? 8 : 0) + ((lane >= 16) ? 8 : 0);
  int N = j * 16 + (lane & 15);
  Wt[tid] = (_Float16)W[K * (2 * DD) + N];
}

// ---------------------------------------------------------------------------
// Kernel 2: LayerNorm over D=128, one wave (32 lanes) per row, f16 output.
// Each lane owns 4 consecutive floats (float4); wave reduction via shfl_xor.
// ---------------------------------------------------------------------------
__global__ void ln_kernel(const float* __restrict__ x,
                          const float* __restrict__ gamma,
                          const float* __restrict__ beta,
                          _Float16* __restrict__ xn) {
  int row  = blockIdx.x * 8 + (threadIdx.x >> 5);     // 8 waves / block
  int lane = threadIdx.x & 31;
  const float4* xr = (const float4*)(x + (size_t)row * DD);
  float4 v = xr[lane];
  float s  = v.x + v.y + v.z + v.w;
  float ss = v.x * v.x + v.y * v.y + v.z * v.z + v.w * v.w;
#pragma unroll
  for (int m = 16; m >= 1; m >>= 1) {
    s  += __shfl_xor(s,  m, 32);
    ss += __shfl_xor(ss, m, 32);
  }
  float mu   = s * (1.0f / DD);
  float var  = ss * (1.0f / DD) - mu * mu;
  float rstd = rsqrtf(var + LN_EPS);
  float4 g  = ((const float4*)gamma)[lane];
  float4 be = ((const float4*)beta)[lane];
  v4h o;
  o[0] = (_Float16)((v.x - mu) * rstd * g.x + be.x);
  o[1] = (_Float16)((v.y - mu) * rstd * g.y + be.y);
  o[2] = (_Float16)((v.z - mu) * rstd * g.z + be.z);
  o[3] = (_Float16)((v.w - mu) * rstd * g.w + be.w);
  ((v4h*)(xn + (size_t)row * DD))[lane] = o;
}

// ---------------------------------------------------------------------------
// Kernel 3: WMMA GEMM (M=131072, K=128, N=256) fused with the log-space
// activation.
//   * One wave owns one 16-row M-tile: A (16x128 f16) is loaded ONCE into
//     32 VGPRs and reused across all 8 (gate, hidden) N-tile pairs
//     -> 64 v_wmma_f32_16x16x32_f16 per wave, A traffic cut 8x.
//   * The full packed B (Wt, 64 KB) is staged once per workgroup into LDS;
//     WMMA B operands come from ds_load_b128 (conflict-free reuse, frees
//     WGP$/L2 bandwidth for the streaming A reads).
//   * Gate tile j and hidden tile j+8 are computed together so the
//     (gate[d], hidden[d]) pair for the activation is register resident.
// A layout (16-bit 16x32): lane 0-15 row M=lane, halves = K {0..7,16..23};
// lane 16-31 same row, halves = K {8..15,24..31} -> two 16B loads per k-step.
// ---------------------------------------------------------------------------
__global__ void gemm_act_kernel(const _Float16* __restrict__ xn,
                                const _Float16* __restrict__ Wt,
                                const float* __restrict__ bias,
                                float* __restrict__ lc,
                                float* __restrict__ lv) {
  __shared__ __align__(32) _Float16 sW[32768];        // 64 KB: entire packed W

  // Stage Wt -> LDS (128 threads x 16 x 32B = 64 KB)
  {
    const v16h* g = (const v16h*)Wt;
    v16h*       s = (v16h*)sW;
#pragma unroll
    for (int i = 0; i < 16; ++i)
      s[threadIdx.x + i * 128] = g[threadIdx.x + i * 128];
  }
  __syncthreads();

  int wid   = (blockIdx.x * blockDim.x + threadIdx.x) >> 5;  // == mtile
  int lane  = threadIdx.x & 31;
  int mtile = wid;

  int row = mtile * 16 + (lane & 15);
  const v8h* arow = (const v8h*)(xn + (size_t)row * DD); // 16 chunks of 8 halves
  int sel = lane >> 4;                                   // 0: K 0..7/16..23, 1: +8

  V16H A[4];
#pragma unroll
  for (int kk = 0; kk < 4; ++kk) {
    A[kk].h[0] = arow[kk * 4 + sel];       // K = kk*32 + sel*8 ..
    A[kk].h[1] = arow[kk * 4 + 2 + sel];   // K = kk*32 + 16 + sel*8 ..
  }

  const v16h* bs = (const v16h*)sW;
  int rbase = mtile * 16 + (sel ? 8 : 0);

  for (int j = 0; j < 8; ++j) {            // gate tile j, hidden tile j+8
    v8f cg = {};
    v8f ch = {};
#pragma unroll
    for (int kk = 0; kk < 4; ++kk) {
      v16h Bg = bs[(j * 4 + kk) * 32 + lane];          // ds_load_b128 x2
      v16h Bh = bs[((j + 8) * 4 + kk) * 32 + lane];
      cg = __builtin_amdgcn_wmma_f32_16x16x32_f16(false, A[kk].v, false, Bg,
                                                  (short)0, cg, false, false);
      ch = __builtin_amdgcn_wmma_f32_16x16x32_f16(false, A[kk].v, false, Bh,
                                                  (short)0, ch, false, false);
    }

    // C/D layout: lanes 0-15 -> N=lane, VGPR m -> M=m; lanes 16-31 -> M=8+m.
    int d  = j * 16 + (lane & 15);
    float gb = bias[d];
    float hb = bias[d + DD];
#pragma unroll
    for (int m = 0; m < 8; ++m) {
      float g = cg[m] + gb;
      float h = ch[m] + hb;
      float lcv = -softplusf(g);                 // log_coeffs
      float lzv = -softplusf(-g);                // log_z
      float glv = (h >= 0.0f) ? logf(h + 0.5f)   // g_act_log
                              : (h - log1pf(expf(h)));
      size_t idx = (size_t)(rbase + m) * DD + d;
      lc[idx] = lcv;
      lv[idx] = lzv + glv;                       // log_values
    }
  }
}

// ---------------------------------------------------------------------------
// Kernel 4: sequential scan over S per (b,t,d) channel.
// a   = running cumsum of log_coeffs (a_star, position s+1)
// L   = running logsumexp of (log_values - a_star), seeded with log(prev_h)
// out = exp(a + L).  Lanes map to consecutive d -> coalesced 128B accesses.
// Stride per step is 8 KB (new cacheline every step) so prefetch a few
// iterations ahead (global_prefetch_b8; speculative, OOB silently dropped).
// ---------------------------------------------------------------------------
__global__ void scan_kernel(const float* __restrict__ lc,
                            const float* __restrict__ lv,
                            const float* __restrict__ prev_hidden,
                            float* __restrict__ out,
                            float* __restrict__ next_prev) {
  int tid = blockIdx.x * blockDim.x + threadIdx.x;   // 0..B*T*D-1 = 32767
  int b   = tid >> 11;                               // / (T*D = 2048)
  int r   = tid & 2047;                              // t*D + d
  float a = 0.0f;
  float L = logf(prev_hidden[tid]);                  // (B,1,T,D) flat == tid
  size_t base = (size_t)b * SS * (TT * DD) + r;
  float last = 0.0f;
  for (int s = 0; s < SS; ++s) {
    size_t idx = base + (size_t)s * (TT * DD);
    size_t pf  = idx + (size_t)8 * (TT * DD);        // 8 steps ahead
    __builtin_prefetch(lc + pf, 0, 0);
    __builtin_prefetch(lv + pf, 0, 0);
    a += lc[idx];
    float v  = lv[idx] - a;
    float mx = fmaxf(L, v);
    L = mx + log1pf(expf(-fabsf(L - v)));            // logaddexp
    last = expf(a + L);
    out[idx] = last;
  }
  next_prev[tid] = last;                             // out[:, -1:]
}

// ---------------------------------------------------------------------------
extern "C" void kernel_launch(void* const* d_in, const int* in_sizes, int n_in,
                              void* d_out, int out_size, void* d_ws, size_t ws_size,
                              hipStream_t stream) {
  const float* x           = (const float*)d_in[0];
  const float* prev_hidden = (const float*)d_in[1];
  const float* gamma       = (const float*)d_in[2];
  const float* beta        = (const float*)d_in[3];
  const float* W           = (const float*)d_in[4];
  const float* bias        = (const float*)d_in[5];

  // Workspace carve-up (all offsets 256B-aligned):
  //   xn : MM*DD f16  = 33,554,432 B
  //   Wt : 32768 f16  =     65,536 B
  //   lc : MM*DD f32  = 67,108,864 B
  //   lv : MM*DD f32  = 67,108,864 B
  char* ws = (char*)d_ws;
  _Float16* xn = (_Float16*)ws;
  _Float16* Wt = (_Float16*)(ws + (size_t)33554432);
  float*    lc = (float*)(ws + (size_t)33554432 + 65536);
  float*    lv = (float*)(ws + (size_t)33554432 + 65536 + (size_t)MM * DD * 4);

  float* out       = (float*)d_out;
  float* next_prev = out + (size_t)MM * DD;

  // 1) Pack W into WMMA B layout (32768 elems)
  wpack_kernel<<<128, 256, 0, stream>>>(W, Wt);
  // 2) LayerNorm -> f16 (131072 rows, 8 waves/block)
  ln_kernel<<<MM / 8, 256, 0, stream>>>(x, gamma, beta, xn);
  // 3) WMMA GEMM + activation: 8192 M-tile waves, 4 waves/block, LDS-staged B
  gemm_act_kernel<<<MM / 16 / 4, 128, 0, stream>>>(xn, Wt, bias, lc, lv);
  // 4) Sequential scan over S (32768 channels)
  scan_kernel<<<(BB * TT * DD) / 256, 256, 0, stream>>>(lc, lv, prev_hidden,
                                                        out, next_prev);
}
